// MultiHeadAttention_22763326668812
// MI455X (gfx1250) — matrix operand order repaired and checked by reading, not compiled, not run on hardware
//
#include <hip/hip_runtime.h>
#include <hip/hip_bf16.h>

// ---------------------------------------------------------------------------
// Multi-head attention forward on gfx1250 (CDNA5)
//   bf16 WMMA (16x16x32) everywhere, fp32 accumulate, flash-style attention,
//   async global->LDS staging (ASYNCcnt) when the toolchain exposes it.
// Pipeline:
//   cvt:   q/k/v f32 -> bf16 (reused buffer), weights f32 -> bf16 transposed
//   gemm:  Q/K projections -> bf16 [B*S,512]; V projection -> bf16 transposed
//          per head [ (b*H+h)*64+d ][ s ]  (so PV B-frags are contiguous-k)
//   flash: streaming softmax over 32-key chunks, QK^T and PV via WMMA
//   gemm:  attn @ wo + bo -> f32 output
// ---------------------------------------------------------------------------

typedef __attribute__((ext_vector_type(16))) __bf16 v16bf;
typedef __attribute__((ext_vector_type(8)))  float  v8f;
typedef int v4i __attribute__((vector_size(16)));

#define D_MODEL   512
#define NUM_HEADS 8
#define D_HEAD    64
#define BATCH     2

// float -> bf16, round-to-nearest-even
__device__ __forceinline__ __bf16 f2bf(float f) {
  unsigned u = __float_as_uint(f);
  u += 0x7FFFu + ((u >> 16) & 1u);
  unsigned short h = (unsigned short)(u >> 16);
  __bf16 r;
  __builtin_memcpy(&r, &h, sizeof(r));
  return r;
}

// ---------------- async global->LDS (CDNA5) with compile-safe fallback ------
#if __has_builtin(__builtin_amdgcn_global_load_async_to_lds_b128) && \
    __has_builtin(__builtin_amdgcn_s_wait_asynccnt)
#define HAVE_ASYNC_LDS 1
__device__ __forceinline__ void async_b128(const __bf16* g, __bf16* l) {
  __builtin_amdgcn_global_load_async_to_lds_b128(
      (__attribute__((address_space(1))) v4i*)g,
      (__attribute__((address_space(3))) v4i*)l, 0, 0);
}
__device__ __forceinline__ void async_wait0() {
  __builtin_amdgcn_s_wait_asynccnt(0);
}
#else
#define HAVE_ASYNC_LDS 0
#endif

// A-operand K index (contiguous order, matches B fills):
// lane half kh: element e holds K = kh*16 + e
__device__ __forceinline__ int a_kidx(int kh, int e) {
  return kh * 16 + e;
}

// A tile stored row-major [16][ld] bf16 -> one contiguous 16-element run/lane
__device__ __forceinline__ v16bf load_a(const __bf16* base, int ld) {
  const int lane = threadIdx.x & 31;
  const int m = lane & 15, kh = lane >> 4;
  v16bf a;
#pragma unroll
  for (int e = 0; e < 16; ++e) a[e] = base[m * ld + a_kidx(kh, e)];
  return a;
}

// B tile stored n-major [16][ld] (element (k,n) at base[n*ld + k]):
// lane n = lane&15, K = (lane>>4)*16 + e  -> 16 contiguous bf16 per lane
__device__ __forceinline__ v16bf load_b_nk(const __bf16* base, int ld) {
  const int lane = threadIdx.x & 31;
  const int n = lane & 15, kh = lane >> 4;
  v16bf b;
#pragma unroll
  for (int e = 0; e < 16; ++e) b[e] = base[n * ld + kh * 16 + e];
  return b;
}

__device__ __forceinline__ v8f bwmma(v16bf a, v16bf b, v8f c) {
  return __builtin_amdgcn_wmma_f32_16x16x32_bf16(
      false, a, false, b, (short)0, c, false, false);
}

// ---------------------------------------------------------------------------
// Elementwise f32 -> bf16
// ---------------------------------------------------------------------------
__global__ __launch_bounds__(256) void cvt_bf16_kernel(
    const float* __restrict__ s, __bf16* __restrict__ d, int n) {
  int i0 = (blockIdx.x * 256 + threadIdx.x) * 4;
  if (i0 + 3 < n) {
    float4 f = *(const float4*)(s + i0);
    d[i0 + 0] = f2bf(f.x);
    d[i0 + 1] = f2bf(f.y);
    d[i0 + 2] = f2bf(f.z);
    d[i0 + 3] = f2bf(f.w);
  }
}

// Weight f32 [k][n] -> bf16 transposed [n][k]  (512x512)
__global__ __launch_bounds__(256) void cvt_wt_kernel(
    const float* __restrict__ w, __bf16* __restrict__ wt) {
  int idx = blockIdx.x * 256 + threadIdx.x;
  int k = idx >> 9, n = idx & 511;
  wt[(size_t)n * D_MODEL + k] = f2bf(w[idx]);
}

// ---------------------------------------------------------------------------
// GEMM: Y[M,512] = X[M,512](bf16) @ W(bf16, pre-transposed [n][k]) + bias
// Block tile 64x64, 4 waves (one 16-row A slice each), K chunks of 32.
// OUT_MODE: 0 = bf16 row-major, 1 = f32 row-major, 2 = bf16 per-head
//           transposed [ (b*H+h)*64+d ][ s ]   (for the V projection)
// ---------------------------------------------------------------------------
template <int OUT_MODE>
__global__ __launch_bounds__(128) void gemm_bias_kernel(
    const __bf16* __restrict__ X, const __bf16* __restrict__ Wt,
    const float* __restrict__ bias, void* __restrict__ Yv, int S) {
  __shared__ __bf16 Xl[64 * 32];   // [m][k]
  __shared__ __bf16 Wl[64 * 32];   // [n][k]
  const int tid  = threadIdx.x;
  const int lane = tid & 31, wave = tid >> 5;
  const int m0 = blockIdx.x * 64;
  const int n0 = blockIdx.y * 64;

  v8f acc[4] = {};

  for (int kc = 0; kc < D_MODEL; kc += 32) {
#if HAVE_ASYNC_LDS
#pragma unroll
    for (int it = 0; it < 2; ++it) {
      int idx = tid + it * 128;          // 0..255: row = idx>>2, 16B seg = idx&3
      int r = idx >> 2, sg = idx & 3;
      async_b128(&X[(size_t)(m0 + r) * D_MODEL + kc + sg * 8], &Xl[r * 32 + sg * 8]);
      async_b128(&Wt[(size_t)(n0 + r) * D_MODEL + kc + sg * 8], &Wl[r * 32 + sg * 8]);
    }
    async_wait0();
#else
#pragma unroll
    for (int i = 0; i < 16; ++i) {
      int e = tid + i * 128;             // e = r*32 + c
      int r = e >> 5, c = e & 31;
      Xl[e] = X[(size_t)(m0 + r) * D_MODEL + kc + c];
      Wl[e] = Wt[(size_t)(n0 + r) * D_MODEL + kc + c];
    }
#endif
    __syncthreads();

    v16bf a = load_a(&Xl[wave * 16 * 32], 32);
#pragma unroll
    for (int nt = 0; nt < 4; ++nt) {
      v16bf b = load_b_nk(&Wl[nt * 16 * 32], 32);
      acc[nt] = bwmma(a, b, acc[nt]);
    }
    __syncthreads();
  }

  const int nl = lane & 15, mb = (lane >> 4) * 8;
#pragma unroll
  for (int nt = 0; nt < 4; ++nt) {
    float bv = bias[n0 + nt * 16 + nl];
#pragma unroll
    for (int r = 0; r < 8; ++r) {
      int row = m0 + wave * 16 + mb + r;
      int col = n0 + nt * 16 + nl;
      float v = acc[nt][r] + bv;
      if (OUT_MODE == 0) {
        ((__bf16*)Yv)[(size_t)row * D_MODEL + col] = f2bf(v);
      } else if (OUT_MODE == 1) {
        ((float*)Yv)[(size_t)row * D_MODEL + col] = v;
      } else {  // per-head transposed V: [ (b*H+h)*64+d ][ s ], contiguous in r
        int h = col >> 6, d = col & 63;
        int bb = row / S;
        int s = row - bb * S;
        ((__bf16*)Yv)[((size_t)(bb * NUM_HEADS + h) * D_HEAD + d) * S + s] = f2bf(v);
      }
    }
  }
}

// ---------------------------------------------------------------------------
// Flash attention: block = 64 queries of one (b,h); 4 waves x 16 queries each.
// K chunk staged [key][dh] (score B-frags contiguous-k),
// V chunk staged [dh][key] from pre-transposed Vt (PV B-frags contiguous-k).
// ---------------------------------------------------------------------------
__global__ __launch_bounds__(128) void flash_attn_kernel(
    const __bf16* __restrict__ Qp, const __bf16* __restrict__ Kp,
    const __bf16* __restrict__ Vt, __bf16* __restrict__ Op, int S) {
  __shared__ __bf16 Kl[32 * 64];       // [key][dh]
  __shared__ __bf16 Vl[64 * 32];       // [dh][key]
  __shared__ __bf16 Pl[4][16 * 32];    // per-wave P transpose buffer [q][key]

  const int tid  = threadIdx.x;
  const int lane = tid & 31, wave = tid >> 5;
  const int b = blockIdx.x / NUM_HEADS, h = blockIdx.x % NUM_HEADS;
  const int hc = h * D_HEAD;
  const int q0 = blockIdx.y * 64;
  const size_t grow0 = (size_t)b * S + q0 + wave * 16;
  const size_t kbase = (size_t)b * S;
  const size_t vtbase = (size_t)(b * NUM_HEADS + h) * D_HEAD * S;

  const int m = lane & 15, kh = lane >> 4;
  const int nl = lane & 15, mb = (lane >> 4) * 8;

  // Q fragments for this wave's 16 rows (dh=64 -> two K=32 chunks)
  v16bf aq[2];
#pragma unroll
  for (int kd = 0; kd < 2; ++kd) {
#pragma unroll
    for (int e = 0; e < 16; ++e)
      aq[kd][e] = Qp[(grow0 + m) * D_MODEL + hc + kd * 32 + a_kidx(kh, e)];
  }

  float mrun[8], lrun[8];
  v8f oacc[4] = {};
#pragma unroll
  for (int r = 0; r < 8; ++r) { mrun[r] = -1e30f; lrun[r] = 0.0f; }

  const float scale = 0.125f;  // 1/sqrt(64)

  for (int kc = 0; kc < S; kc += 32) {
#if HAVE_ASYNC_LDS
#pragma unroll
    for (int it = 0; it < 2; ++it) {
      int idx = tid + it * 128;            // 0..255
      int kr = idx >> 3, sgk = idx & 7;    // K: 32 rows x 8 segs of 8 bf16
      async_b128(&Kp[(kbase + kc + kr) * D_MODEL + hc + sgk * 8],
                 &Kl[kr * 64 + sgk * 8]);
      int dv = idx >> 2, sgv = idx & 3;    // V: 64 rows x 4 segs of 8 bf16
      async_b128(&Vt[vtbase + (size_t)dv * S + kc + sgv * 8],
                 &Vl[dv * 32 + sgv * 8]);
    }
    async_wait0();
#else
#pragma unroll
    for (int i = 0; i < 16; ++i) {
      int e = tid + i * 128;
      int kr = e >> 6, ck = e & 63;
      Kl[e] = Kp[(kbase + kc + kr) * D_MODEL + hc + ck];
      int dv = e >> 5, cv = e & 31;
      Vl[e] = Vt[vtbase + (size_t)dv * S + kc + cv];
    }
#endif
    __syncthreads();

    // scores: 16 queries x 32 keys (two 16-key N tiles, two dh K-chunks)
    v8f s0 = {}, s1 = {};
#pragma unroll
    for (int kd = 0; kd < 2; ++kd) {
      v16bf b0 = load_b_nk(&Kl[0 * 64 + kd * 32], 64);   // keys 0..15
      v16bf b1 = load_b_nk(&Kl[16 * 64 + kd * 32], 64);  // keys 16..31
      s0 = bwmma(aq[kd], b0, s0);
      s1 = bwmma(aq[kd], b1, s1);
    }

    // online softmax: lane holds rows mb..mb+7 at key columns nl and nl+16
    v8f p0, p1;
#pragma unroll
    for (int r = 0; r < 8; ++r) {
      float a0 = s0[r] * scale;
      float a1 = s1[r] * scale;
      float cm = fmaxf(a0, a1);
#pragma unroll
      for (int msk = 1; msk < 16; msk <<= 1)
        cm = fmaxf(cm, __shfl_xor(cm, msk, 32));
      float mnew = fmaxf(mrun[r], cm);
      float corr = __expf(mrun[r] - mnew);
      float e0 = __expf(a0 - mnew);
      float e1 = __expf(a1 - mnew);
      p0[r] = e0; p1[r] = e1;
      float rs = e0 + e1;
#pragma unroll
      for (int msk = 1; msk < 16; msk <<= 1)
        rs += __shfl_xor(rs, msk, 32);
      lrun[r] = lrun[r] * corr + rs;
      mrun[r] = mnew;
#pragma unroll
      for (int nt = 0; nt < 4; ++nt) oacc[nt][r] *= corr;
    }

    // transpose P (C-layout) -> A-layout via per-wave LDS buffer
#pragma unroll
    for (int r = 0; r < 8; ++r) {
      Pl[wave][(mb + r) * 32 + nl]      = f2bf(p0[r]);
      Pl[wave][(mb + r) * 32 + 16 + nl] = f2bf(p1[r]);
    }
    __syncthreads();

    v16bf ap = load_a(&Pl[wave][0], 32);
#pragma unroll
    for (int nt = 0; nt < 4; ++nt) {
      v16bf bv = load_b_nk(&Vl[nt * 16 * 32], 32);
      oacc[nt] = bwmma(ap, bv, oacc[nt]);
    }
    __syncthreads();
  }

  // normalize and store bf16 attention output [B*S, 512]
#pragma unroll
  for (int nt = 0; nt < 4; ++nt) {
#pragma unroll
    for (int r = 0; r < 8; ++r) {
      float inv = 1.0f / lrun[r];
      Op[(grow0 + mb + r) * D_MODEL + hc + nt * 16 + nl] = f2bf(oacc[nt][r] * inv);
    }
  }
}

extern "C" void kernel_launch(void* const* d_in, const int* in_sizes, int n_in,
                              void* d_out, int out_size, void* d_ws, size_t ws_size,
                              hipStream_t stream) {
  const float* q  = (const float*)d_in[0];
  const float* k  = (const float*)d_in[1];
  const float* v  = (const float*)d_in[2];
  const float* wq = (const float*)d_in[3];
  const float* bq = (const float*)d_in[4];
  const float* wk = (const float*)d_in[5];
  const float* bk = (const float*)d_in[6];
  const float* wv = (const float*)d_in[7];
  const float* bv = (const float*)d_in[8];
  const float* wo = (const float*)d_in[9];
  const float* bo = (const float*)d_in[10];

  const int M = in_sizes[0] / D_MODEL;   // B*S = 8192
  const int S = M / BATCH;               // 4096
  const size_t seg = (size_t)M * D_MODEL;          // 4M bf16 elements
  const size_t wseg = (size_t)D_MODEL * D_MODEL;   // 256K bf16 elements

  __bf16* Xb  = (__bf16*)d_ws;        // reused for q, k, v conversions
  __bf16* Qp  = Xb + seg;
  __bf16* Kp  = Qp + seg;
  __bf16* Vtp = Kp + seg;             // per-head transposed V
  __bf16* Ap  = Vtp + seg;
  __bf16* Wqt = Ap + seg;
  __bf16* Wkt = Wqt + wseg;
  __bf16* Wvt = Wkt + wseg;
  __bf16* Wot = Wvt + wseg;

  const int nAct = M * D_MODEL;
  dim3 cg(nAct / 1024);                 // cvt: 4 elems/thread, 256 thr/block
  dim3 wg(wseg / 256);
  dim3 gg(M / 64, D_MODEL / 64);
  dim3 fg(BATCH * NUM_HEADS, S / 64);

  // weights -> bf16 transposed
  cvt_wt_kernel<<<wg, 256, 0, stream>>>(wq, Wqt);
  cvt_wt_kernel<<<wg, 256, 0, stream>>>(wk, Wkt);
  cvt_wt_kernel<<<wg, 256, 0, stream>>>(wv, Wvt);
  cvt_wt_kernel<<<wg, 256, 0, stream>>>(wo, Wot);

  // Q projection
  cvt_bf16_kernel<<<cg, 256, 0, stream>>>(q, Xb, nAct);
  gemm_bias_kernel<0><<<gg, 128, 0, stream>>>(Xb, Wqt, bq, Qp, S);
  // K projection
  cvt_bf16_kernel<<<cg, 256, 0, stream>>>(k, Xb, nAct);
  gemm_bias_kernel<0><<<gg, 128, 0, stream>>>(Xb, Wkt, bk, Kp, S);
  // V projection (per-head transposed output)
  cvt_bf16_kernel<<<cg, 256, 0, stream>>>(v, Xb, nAct);
  gemm_bias_kernel<2><<<gg, 128, 0, stream>>>(Xb, Wvt, bv, Vtp, S);

  // attention
  flash_attn_kernel<<<fg, 128, 0, stream>>>(Qp, Kp, Vtp, Ap, S);

  // output projection (f32 out)
  gemm_bias_kernel<1><<<gg, 128, 0, stream>>>(Ap, Wot, bo, d_out, S);
}
